// CustomMHA_31018253812045
// MI455X (gfx1250) — compile-verified, matrix operand order checked by reading, not run on hardware
//
#include <hip/hip_runtime.h>
#include <hip/hip_bf16.h>
#include <stdint.h>

typedef __attribute__((ext_vector_type(16))) _Float16 v16h;
typedef __attribute__((ext_vector_type(8)))  _Float16 v8h;
typedef __attribute__((ext_vector_type(4)))  _Float16 v4h;
typedef __attribute__((ext_vector_type(8)))  float    v8f;

#define NH     16
#define DH     64
#define SEQ    2048
#define BATCH  2
#define DMODEL 1024

// 16x32 (rows x K) f16 A/B fragment from a row-major f16 matrix (global or LDS).
// ISA 7.12.2: lane<16 -> K {0..7,16..23}, lane>=16 -> K {8..15,24..31}, row = lane&15.
// Two 16-byte loads per lane -> global_load_b128 / ds_load_b128.
__device__ __forceinline__ v16h load_frag_h(const _Float16* base, int ld) {
  const int lane = threadIdx.x & 31;
  const int r = lane & 15, hi = lane >> 4;
  const _Float16* p = base + r * ld + 8 * hi;
  v8h a = *(const v8h*)(p);
  v8h b = *(const v8h*)(p + 16);
  return __builtin_shufflevector(a, b, 0, 1, 2, 3, 4, 5, 6, 7, 8, 9, 10, 11, 12, 13, 14, 15);
}

// One-time fp32 -> f16 pack (n4 = elements/4).
__global__ __launch_bounds__(256) void cvt_f32_f16(const float* __restrict__ s,
                                                   _Float16* __restrict__ d, int n4) {
  const int i = blockIdx.x * 256 + threadIdx.x;
  if (i < n4) {
    float4 v = ((const float4*)s)[i];
    v4h h = {(_Float16)v.x, (_Float16)v.y, (_Float16)v.z, (_Float16)v.w};
    ((v4h*)d)[i] = h;
  }
}

// C = A[M,K] @ W[N,K]^T, f16 operands, f32 accumulate.
// 4 waves/block, wave tile 16x64, block tile 64x64.  The shared W tile
// (64 cols x 32 K = 4KB) is double-buffered in LDS via async copies so the
// HBM->LDS DMA of tile k+1 overlaps the WMMAs on tile k.
// MODE 0: store f32 C.  MODE 1: QKV epilogue scatter (f32 K/V outputs + f16 copies,
// with the 1/sqrt(dh) attention scale folded into Q).
template <int MODE>
__global__ __launch_bounds__(128) void gemm_wmma(
    const _Float16* A, const _Float16* W, float* C,
    _Float16* q_h, float* k_out, _Float16* k_h,
    float* v_out, _Float16* vt_h, int M, int N, int K) {
  __shared__ _Float16 wtile[2][64 * 32];
  const int wid = threadIdx.x >> 5;
  const int m0 = blockIdx.x * 64 + wid * 16;
  const int n0 = blockIdx.y * 64;
  const int nk = K / 32;

  auto issue_tile = [&](int kb, int buf) {
    const int c0 = threadIdx.x * 2;       // 256 x 16B chunks, 2 per thread
#pragma unroll
    for (int u = 0; u < 2; ++u) {
      const int c = c0 + u;
      const int row = c >> 2, off = (c & 3) * 8;                    // halves
      unsigned lo = (unsigned)(uintptr_t)&wtile[buf][row * 32 + off];  // LDS byte addr
      const void* g = (const void*)(W + (long)(n0 + row) * K + kb * 32 + off);
      asm volatile("global_load_async_to_lds_b128 %0, %1, off"
                   :: "v"(lo), "v"(g) : "memory");
    }
  };

  v8f acc[4] = {};
  issue_tile(0, 0);
  for (int kb = 0; kb < nk; ++kb) {
    const bool hasnext = (kb + 1) < nk;
    if (hasnext) {
      issue_tile(kb + 1, (kb + 1) & 1);
      asm volatile("s_wait_asynccnt 0x2" ::: "memory");  // prev tile landed; next in flight
    } else {
      asm volatile("s_wait_asynccnt 0x0" ::: "memory");
    }
    __syncthreads();

    v16h a = load_frag_h(A + (long)m0 * K + kb * 32, K);
#pragma unroll
    for (int t = 0; t < 4; ++t) {
      v16h b = load_frag_h(&wtile[kb & 1][16 * t * 32], 32);
      acc[t] = __builtin_amdgcn_wmma_f32_16x16x32_f16(
          false, a, false, b, (short)0, acc[t], false, false);
    }
    __syncthreads();   // reads done before this buffer is overwritten (next iter)
  }

  const int lane = threadIdx.x & 31;
  const int nl = lane & 15, hi = lane >> 4;
#pragma unroll
  for (int t = 0; t < 4; ++t) {
#pragma unroll
    for (int i = 0; i < 8; ++i) {
      const int m = m0 + i + 8 * hi;       // C layout: row = vgpr + 8*hi
      const int n = n0 + 16 * t + nl;      //           col = lane & 15
      const float v = acc[t][i];
      if (MODE == 0) {
        C[(long)m * N + n] = v;
      } else {
        const int b = m / SEQ, s = m % SEQ;
        const int which = n >> 10;         // 0=Q 1=K 2=V
        const int cm = n & (DMODEL - 1);
        const int h = cm >> 6, d = cm & (DH - 1);
        const int bh = b * NH + h;
        const long idx = ((long)bh * SEQ + s) * DH + d;
        if (which == 0)      q_h[idx] = (_Float16)(v * 0.125f);  // fold 1/sqrt(64)
        else if (which == 1) { k_out[idx] = v; k_h[idx] = (_Float16)v; }
        else                 { v_out[idx] = v;
                               vt_h[((long)bh * DH + d) * SEQ + s] = (_Float16)v; }
      }
    }
  }
}

// One 32-key flash-attention block.  MASKED=true only for the diagonal block.
template <bool MASKED>
__device__ __forceinline__ void attn_block(
    int k0, int q0, int nl, int hi,
    const _Float16* __restrict__ kbh,   // k_h  + bh*SEQ*DH
    const _Float16* __restrict__ vtb,   // vt_h + bh*DH*SEQ
    const v16h& qa0, const v16h& qa1,
    _Float16* pl, v8f (&o)[4], float (&mrow)[8], float (&lrow)[8]) {
  float pnew[2][8], rmax[8];
#pragma unroll
  for (int i = 0; i < 8; ++i) rmax[i] = -INFINITY;

#pragma unroll
  for (int sub = 0; sub < 2; ++sub) {    // two 16-key sub-tiles
    const _Float16* kbase = kbh + (long)(k0 + 16 * sub) * DH;
    const v16h b0 = load_frag_h(kbase, DH);
    const v16h b1 = load_frag_h(kbase + 32, DH);
    v8f s = {};
    s = __builtin_amdgcn_wmma_f32_16x16x32_f16(false, qa0, false, b0, (short)0, s, false, false);
    s = __builtin_amdgcn_wmma_f32_16x16x32_f16(false, qa1, false, b1, (short)0, s, false, false);
#pragma unroll
    for (int i = 0; i < 8; ++i) {
      float sv = s[i];                   // scale pre-folded into Q
      if (MASKED) {
        const int key = k0 + 16 * sub + nl;
        const int qrow = q0 + i + 8 * hi;
        sv = (key <= qrow) ? sv : -INFINITY;
      }
      pnew[sub][i] = sv;
      rmax[i] = fmaxf(rmax[i], sv);
    }
  }
#pragma unroll
  for (int i = 0; i < 8; ++i) {          // row max over the 16-lane group
    float v = rmax[i];
#pragma unroll
    for (int off = 1; off < 16; off <<= 1) v = fmaxf(v, __shfl_xor(v, off, 32));
    rmax[i] = v;
  }
  float alpha[8];
#pragma unroll
  for (int i = 0; i < 8; ++i) {
    const float mn = fmaxf(mrow[i], rmax[i]);
    alpha[i] = __expf(mrow[i] - mn);
    mrow[i] = mn;
  }
  float rsum[8];
#pragma unroll
  for (int i = 0; i < 8; ++i) rsum[i] = 0.f;
#pragma unroll
  for (int sub = 0; sub < 2; ++sub)
#pragma unroll
    for (int i = 0; i < 8; ++i) {
      const float p = __expf(pnew[sub][i] - mrow[i]);
      rsum[i] += p;
      pl[(i + 8 * hi) * 32 + 16 * sub + nl] = (_Float16)p;   // C-layout -> LDS tile
    }
#pragma unroll
  for (int i = 0; i < 8; ++i) {
    float v = rsum[i];
#pragma unroll
    for (int off = 1; off < 16; off <<= 1) v += __shfl_xor(v, off, 32);
    lrow[i] = lrow[i] * alpha[i] + v;
  }
#pragma unroll
  for (int t = 0; t < 4; ++t)
#pragma unroll
    for (int i = 0; i < 8; ++i) o[t][i] *= alpha[i];

  asm volatile("s_wait_dscnt 0x0" ::: "memory");   // P stores visible before re-read
  const v16h pfrag = load_frag_h(pl, 32);          // P in A-fragment layout
#pragma unroll
  for (int t = 0; t < 4; ++t) {                    // O += P[16x32] @ V[32keys x 16dh]
    const v16h vb = load_frag_h(vtb + (long)(16 * t) * SEQ + k0, SEQ);
    o[t] = __builtin_amdgcn_wmma_f32_16x16x32_f16(false, pfrag, false, vb, (short)0, o[t], false, false);
  }
}

// Flash attention: one wave per (bh, 16-query tile); 32-key blocks, online softmax.
__global__ __launch_bounds__(128) void attn_kernel(
    const _Float16* __restrict__ q_h, const _Float16* __restrict__ k_h,
    const _Float16* __restrict__ vt_h, _Float16* __restrict__ o_h) {
  __shared__ _Float16 lds_p[4][16 * 32];
  const int wid = threadIdx.x >> 5;
  const int lane = threadIdx.x & 31;
  const int nl = lane & 15, hi = lane >> 4;
  const int job = blockIdx.x * 4 + wid;
  const int bh = job >> 7;                 // 128 query tiles per head
  const int q0 = (job & 127) * 16;
  const int b = bh >> 4, h = bh & 15;

  const _Float16* qbase = q_h + ((long)bh * SEQ + q0) * DH;
  const v16h qa0 = load_frag_h(qbase, DH);        // dh 0..31 (pre-scaled by 1/8)
  const v16h qa1 = load_frag_h(qbase + 32, DH);   // dh 32..63
  const _Float16* kbh = k_h + (long)bh * SEQ * DH;
  const _Float16* vtb = vt_h + (long)bh * DH * SEQ;

  v8f o[4] = {};
  float mrow[8], lrow[8];
#pragma unroll
  for (int i = 0; i < 8; ++i) { mrow[i] = -INFINITY; lrow[i] = 0.f; }

  _Float16* pl = &lds_p[wid][0];
  const int nfull = (q0 + 1) / 32;         // key blocks entirely below the diagonal
  const int nkb = (q0 + 15) / 32 + 1;      // total causal key blocks

  for (int kb = 0; kb < nfull; ++kb)       // hot path: no masking, no selects
    attn_block<false>(kb * 32, q0, nl, hi, kbh, vtb, qa0, qa1, pl, o, mrow, lrow);
  for (int kb = nfull; kb < nkb; ++kb)     // diagonal block only
    attn_block<true>(kb * 32, q0, nl, hi, kbh, vtb, qa0, qa1, pl, o, mrow, lrow);

#pragma unroll
  for (int i = 0; i < 8; ++i) lrow[i] = 1.f / lrow[i];
#pragma unroll
  for (int t = 0; t < 4; ++t)
#pragma unroll
    for (int i = 0; i < 8; ++i) {
      const int s = q0 + i + 8 * hi;
      const int d = h * DH + 16 * t + nl;
      o_h[((long)b * SEQ + s) * DMODEL + d] = (_Float16)(o[t][i] * lrow[i]);
    }
}

extern "C" void kernel_launch(void* const* d_in, const int* in_sizes, int n_in,
                              void* d_out, int out_size, void* d_ws, size_t ws_size,
                              hipStream_t stream) {
  const float* x   = (const float*)d_in[0];   // [B,S,D]
  const float* qkv = (const float*)d_in[1];   // [3D,D]
  const float* wo  = (const float*)d_in[2];   // [D,D]
  float* out = (float*)d_out;
  const long MD = (long)BATCH * SEQ * DMODEL;           // 4,194,304
  float* k_out = out + MD;                              // k_heads [B*H,S,dh] (f32 output)
  float* v_out = out + 2 * MD;                          // v_heads [B*H,S,dh] (f32 output)

  // f16 workspace (halves): ~48 MiB total
  _Float16* wsh   = (_Float16*)d_ws;
  _Float16* x_h   = wsh;                                //  4,194,304
  _Float16* qkv_h = x_h + MD;                           //  3,145,728
  _Float16* wo_h  = qkv_h + 3 * DMODEL * DMODEL;        //  1,048,576
  _Float16* q_h   = wo_h + DMODEL * DMODEL;             //  4,194,304 (pre-scaled by 1/8)
  _Float16* k_h   = q_h + MD;                           //  4,194,304
  _Float16* vt_h  = k_h + MD;                           //  4,194,304  V^T [B*H,dh,S]
  _Float16* o_h   = vt_h + MD;                          //  4,194,304  attn out [B,S,D]

  const int M = BATCH * SEQ;                            // 4096
  dim3 blk128(128), blk256(256);

  cvt_f32_f16<<<dim3((int)(MD / 4 / 256)), blk256, 0, stream>>>(x, x_h, (int)(MD / 4));
  cvt_f32_f16<<<dim3(3 * DMODEL * DMODEL / 4 / 256), blk256, 0, stream>>>(qkv, qkv_h, 3 * DMODEL * DMODEL / 4);
  cvt_f32_f16<<<dim3(DMODEL * DMODEL / 4 / 256), blk256, 0, stream>>>(wo, wo_h, DMODEL * DMODEL / 4);

  gemm_wmma<1><<<dim3(M / 64, (3 * DMODEL) / 64), blk128, 0, stream>>>(
      x_h, qkv_h, nullptr, q_h, k_out, k_h, v_out, vt_h, M, 3 * DMODEL, DMODEL);

  attn_kernel<<<dim3((BATCH * NH * (SEQ / 16)) / 4), blk128, 0, stream>>>(
      q_h, k_h, vt_h, o_h);

  gemm_wmma<0><<<dim3(M / 64, DMODEL / 64), blk128, 0, stream>>>(
      o_h, wo_h, out, nullptr, nullptr, nullptr, nullptr, nullptr, M, DMODEL, DMODEL);
}
